// CrossAttention_601295421830
// MI455X (gfx1250) — matrix operand order repaired and checked by reading, not compiled, not run on hardware
//
#include <hip/hip_runtime.h>

// ---------------------------------------------------------------------------
// CrossAttention for MI455X (gfx1250, wave32, WMMA bf16 16x16x32, f32 accum)
//
//  1) cvt_f32_bf16: K,Q,V [4096,1024] and WK,WQ,WV [1024,1024] f32 -> bf16
//  2) proj_ln_kernel (x3): Y = X*W^T + b, per-head LayerNorm(Dh=64).
//     W k-slices ASYNC-staged into LDS (double buffered, shared by 8 waves);
//     A fragments software-pipelined in registers; last k-step peeled so the
//     steady-state loop body is branch-free.
//     Key/Query written bf16 [B,H,S,64]; Value written TRANSPOSED [B,H,64,S].
//  3) attn_kernel: flash attention, 64 queries/block (4 waves), 64-key tiles.
//     K and V^T tiles ASYNC-staged into LDS (double buffered). Softmax done
//     in place in the f32 score staging; P fragments converted f32->bf16
//     with vector LDS reads. Last key-tile peeled.
// ---------------------------------------------------------------------------

typedef __attribute__((ext_vector_type(16))) __bf16       v16bf;
typedef __attribute__((ext_vector_type(8)))  float        v8f;
typedef __attribute__((ext_vector_type(4)))  float        v4f;
typedef __attribute__((ext_vector_type(4)))  unsigned int u32x4;

union FragBF { v16bf v; u32x4 q[2]; };

#define DIM   1024
#define NHEAD 16
#define HEAD  64
#define SEQ   2048
#define BATCH 2
#define ROWS  (SEQ * BATCH)   // 4096 flattened (s,b) rows

// ---- CDNA5 async global->LDS copy (ASYNCcnt path), with safe fallback ----
typedef int v4i_gcc __attribute__((vector_size(16)));
typedef __attribute__((address_space(1))) v4i_gcc gv4i;
typedef __attribute__((address_space(3))) v4i_gcc lv4i;

__device__ __forceinline__ void async_ld_b128(const void* gp, void* lp) {
#if __has_builtin(__builtin_amdgcn_global_load_async_to_lds_b128)
  __builtin_amdgcn_global_load_async_to_lds_b128((gv4i*)gp, (lv4i*)lp, 0, 0);
#else
  *(u32x4*)lp = *(const u32x4*)gp;   // synchronous fallback
#endif
}

#if __has_builtin(__builtin_amdgcn_s_wait_asynccnt)
#define WAIT_ASYNC(n) __builtin_amdgcn_s_wait_asynccnt(n)
#elif __has_builtin(__builtin_amdgcn_global_load_async_to_lds_b128)
#define WAIT_ASYNC(n) asm volatile("s_wait_asynccnt %0" ::"n"(n) : "memory")
#else
#define WAIT_ASYNC(n) ((void)0)
#endif

// ---------------------------------------------------------------------------
__global__ void cvt_f32_bf16(const float* __restrict__ in,
                             __bf16* __restrict__ out, int n) {
  int i = blockIdx.x * blockDim.x + threadIdx.x;
  if (i < n) out[i] = (__bf16)in[i];
}

// ---------------------------------------------------------------------------
// Projection + bias + per-head LayerNorm.
// Block: 256 threads = 8 waves; wave = 16 rows x 64 cols (one full head).
// Grid: (ROWS/128, NHEAD).
// ---------------------------------------------------------------------------
__global__ __launch_bounds__(256)
void proj_ln_kernel(const __bf16* __restrict__ X,   // [4096,1024]
                    const __bf16* __restrict__ W,   // [1024,1024] W[o][c]
                    const float*  __restrict__ bias,
                    const float*  __restrict__ g,
                    const float*  __restrict__ beta,
                    __bf16* __restrict__ out,
                    int transposed) {
  __shared__ float  sS[8][16][64];                 // per-wave f32 staging
  __shared__ __align__(16) __bf16 sW[2][64][32];   // double-buffered W tile

  const int t    = threadIdx.x;
  const int lane = t & 31;
  const int wid  = t >> 5;
  const int m    = lane & 15;
  const int half = lane >> 4;
  const int aoff = half * 16;   // A-frag K sub-offset (contiguous, matches B)
  const int boff = half * 16;   // B-frag K sub-offset (contiguous, matches A)
  const int rowBase = blockIdx.x * 128 + wid * 16;
  const int head    = blockIdx.y;

  const __bf16* arow = X + (size_t)(rowBase + m) * DIM;

  auto issueW = [&](int kk, int buf) {   // 64x32 W slice: 256 lanes x 16B
    const __bf16* gp =
        W + (size_t)(head * HEAD + (t >> 2)) * DIM + kk + (t & 3) * 8;
    async_ld_b128(gp, &sW[buf][t >> 2][(t & 3) * 8]);
  };
  auto loadA = [&](int kk) {
    FragBF a;
    a.q[0] = *(const u32x4*)(arow + kk + aoff);
    a.q[1] = *(const u32x4*)(arow + kk + aoff + 8);
    return a;
  };

  v8f acc[4];
  const v8f vzero = {0.f, 0.f, 0.f, 0.f, 0.f, 0.f, 0.f, 0.f};
#pragma unroll
  for (int nt = 0; nt < 4; ++nt) acc[nt] = vzero;

  auto gemmStep = [&](int buf, const FragBF& a) {
    FragBF b[4];                      // distinct regs -> batched ds loads
#pragma unroll
    for (int nt = 0; nt < 4; ++nt) {
      const __bf16* bp = &sW[buf][nt * 16 + m][boff];
      b[nt].q[0] = *(const u32x4*)(bp);
      b[nt].q[1] = *(const u32x4*)(bp + 8);
    }
#pragma unroll
    for (int nt = 0; nt < 4; ++nt)
      acc[nt] = __builtin_amdgcn_wmma_f32_16x16x32_bf16(
          false, a.v, false, b[nt].v, (short)0, acc[nt], false, false);
  };

  constexpr int NIT = DIM / 32;   // 32 k-steps
  issueW(0, 0);
  FragBF a = loadA(0);
  for (int it = 0; it < NIT - 1; ++it) {      // branch-free steady state
    FragBF an = loadA((it + 1) * 32);         // pipeline next A
    __builtin_prefetch(arow + it * 32 + 96, 0, 1);
    issueW((it + 1) * 32, (it + 1) & 1);      // prefetch next W slice
    WAIT_ASYNC(1);                            // oldest batch landed
    __syncthreads();
    gemmStep(it & 1, a);
    __syncthreads();
    a = an;
  }
  WAIT_ASYNC(0);                              // peeled last step
  __syncthreads();
  gemmStep((NIT - 1) & 1, a);

  // Bias + stage to LDS as [row][col]
#pragma unroll
  for (int nt = 0; nt < 4; ++nt) {
    const float bval = bias[head * HEAD + nt * 16 + m];
#pragma unroll
    for (int r = 0; r < 8; ++r)
      sS[wid][r + 8 * half][nt * 16 + m] = acc[nt][r] + bval;
  }
  __syncthreads();

  // LayerNorm over Dh=64, one row per lane (lanes 0..15)
  if (lane < 16) {
    float mu = 0.f;
    for (int j = 0; j < 64; ++j) mu += sS[wid][lane][j];
    mu *= (1.0f / 64.0f);
    float var = 0.f;
    for (int j = 0; j < 64; ++j) {
      float d = sS[wid][lane][j] - mu;
      var += d * d;
    }
    var *= (1.0f / 64.0f);
    const float inv = rsqrtf(var + 1e-5f);

    const int gr = rowBase + lane;
    const int srow = gr >> 1;
    const int bcol = gr & 1;
    const size_t bh = (size_t)(bcol * NHEAD + head);
    for (int j = 0; j < 64; ++j) {
      float y = (sS[wid][lane][j] - mu) * inv * g[j] + beta[j];
      size_t idx = transposed ? (bh * HEAD + j) * (size_t)SEQ + srow
                              : (bh * SEQ + srow) * (size_t)HEAD + j;
      out[idx] = (__bf16)y;
    }
  }
}

// ---------------------------------------------------------------------------
// Flash attention. Grid: (SEQ/64, B*H). Block: 128 threads = 4 waves.
// ---------------------------------------------------------------------------
__global__ __launch_bounds__(128)
void attn_kernel(const __bf16* __restrict__ Qn,   // [BH, S, 64]
                 const __bf16* __restrict__ Kn,   // [BH, S, 64]
                 const __bf16* __restrict__ Vt,   // [BH, 64, S]
                 float* __restrict__ out) {       // [S, B, 1024]
  __shared__ float  sS[4][16][64];                 // scores / P / O staging
  __shared__ float  sScale[4][16];
  __shared__ __align__(16) __bf16 sK[2][64][64];   // key tile (keys x d)
  __shared__ __align__(16) __bf16 sV[2][64][64];   // V^T tile (d x keys)

  const int t    = threadIdx.x;
  const int lane = t & 31;
  const int wid  = t >> 5;
  const int m    = lane & 15;
  const int half = lane >> 4;
  const int aoff = half * 16;
  const int boff = half * 16;
  const int bh    = blockIdx.y;
  const int qBase = blockIdx.x * 64 + wid * 16;

  auto issueKV = [&](int kt, int buf) {   // 2 x 8KB tiles: 128 lanes x 4 x 16B
    const int kB = kt * 64;
    const int r  = t >> 1;
    const int c  = (t & 1) * 32;
    const __bf16* kg = Kn + ((size_t)bh * SEQ + kB + r) * HEAD + c;
    const __bf16* vg = Vt + ((size_t)bh * HEAD + r) * SEQ + kB + c;
#pragma unroll
    for (int i = 0; i < 4; ++i) {
      async_ld_b128(kg + i * 8, &sK[buf][r][c + i * 8]);
      async_ld_b128(vg + i * 8, &sV[buf][r][c + i * 8]);
    }
  };

  // Q A-fragments, loaded once (16x64 = two 16x32 frags)
  const __bf16* qrow = Qn + ((size_t)bh * SEQ + qBase + m) * HEAD;
  FragBF qf[2];
#pragma unroll
  for (int j = 0; j < 2; ++j) {
    qf[j].q[0] = *(const u32x4*)(qrow + j * 32 + aoff);
    qf[j].q[1] = *(const u32x4*)(qrow + j * 32 + aoff + 8);
  }

  const v8f vzero = {0.f, 0.f, 0.f, 0.f, 0.f, 0.f, 0.f, 0.f};
  v8f o[4];
#pragma unroll
  for (int nt = 0; nt < 4; ++nt) o[nt] = vzero;
  float mi = -1e30f, li = 0.f;   // online softmax state (lanes 0..15)

  auto tileStep = [&](int buf) {
    // ---- S = Q * K^T (16x64 tile) from LDS key tile ----
    v8f s[4];
#pragma unroll
    for (int nt = 0; nt < 4; ++nt) s[nt] = vzero;
#pragma unroll
    for (int j = 0; j < 2; ++j) {
      FragBF b[4];
#pragma unroll
      for (int nt = 0; nt < 4; ++nt) {
        const __bf16* bp = &sK[buf][nt * 16 + m][j * 32 + boff];
        b[nt].q[0] = *(const u32x4*)(bp);
        b[nt].q[1] = *(const u32x4*)(bp + 8);
      }
#pragma unroll
      for (int nt = 0; nt < 4; ++nt)
        s[nt] = __builtin_amdgcn_wmma_f32_16x16x32_bf16(
            false, qf[j].v, false, b[nt].v, (short)0, s[nt], false, false);
    }
#pragma unroll
    for (int nt = 0; nt < 4; ++nt)
#pragma unroll
      for (int r = 0; r < 8; ++r)
        sS[wid][r + 8 * half][nt * 16 + m] = s[nt][r] * 0.125f;
    __syncthreads();

    // ---- online softmax, in place (lanes 0..15, one row each) ----
    if (lane < 16) {
      float mt = -1e30f;
      for (int j = 0; j < 64; ++j) mt = fmaxf(mt, sS[wid][lane][j]);
      const float mn  = fmaxf(mi, mt);
      const float scl = __expf(mi - mn);
      float sum = 0.f;
      for (int j = 0; j < 64; ++j) {
        float p = __expf(sS[wid][lane][j] - mn);
        sum += p;
        sS[wid][lane][j] = p;
      }
      li = li * scl + sum;
      mi = mn;
      sScale[wid][lane] = scl;
    }
    __syncthreads();

    // ---- rescale O, then O += P * V ----
#pragma unroll
    for (int nt = 0; nt < 4; ++nt)
#pragma unroll
      for (int r = 0; r < 8; ++r)
        o[nt][r] *= sScale[wid][r + 8 * half];

    const float* prow = &sS[wid][m][0];
#pragma unroll
    for (int j = 0; j < 2; ++j) {
      // f32 probabilities -> bf16 A-frag via vector LDS reads
      const v4f c0 = *(const v4f*)(prow + j * 32 + aoff);
      const v4f c1 = *(const v4f*)(prow + j * 32 + aoff + 4);
      const v4f c2 = *(const v4f*)(prow + j * 32 + aoff + 8);
      const v4f c3 = *(const v4f*)(prow + j * 32 + aoff + 12);
      FragBF p;
#pragma unroll
      for (int e = 0; e < 4; ++e) {
        p.v[e]      = (__bf16)c0[e];
        p.v[4 + e]  = (__bf16)c1[e];
        p.v[8 + e]  = (__bf16)c2[e];
        p.v[12 + e] = (__bf16)c3[e];
      }
      FragBF b[4];
#pragma unroll
      for (int nt = 0; nt < 4; ++nt) {
        const __bf16* bp = &sV[buf][nt * 16 + m][j * 32 + boff];
        b[nt].q[0] = *(const u32x4*)(bp);
        b[nt].q[1] = *(const u32x4*)(bp + 8);
      }
#pragma unroll
      for (int nt = 0; nt < 4; ++nt)
        o[nt] = __builtin_amdgcn_wmma_f32_16x16x32_bf16(
            false, p.v, false, b[nt].v, (short)0, o[nt], false, false);
    }
    __syncthreads();   // reads of buf / sS done before next overwrite
  };

  constexpr int NT = SEQ / 64;   // 32 key tiles
  issueKV(0, 0);
  for (int kt = 0; kt < NT - 1; ++kt) {       // branch-free steady state
    issueKV(kt + 1, (kt + 1) & 1);            // 8 async ops per lane
    WAIT_ASYNC(8);                            // previous batch complete
    __syncthreads();
    tileStep(kt & 1);
  }
  WAIT_ASYNC(0);                              // peeled last tile
  __syncthreads();
  tileStep((NT - 1) & 1);

  // ---- epilogue: O / l, write [S, B, DIM] f32 ----
#pragma unroll
  for (int nt = 0; nt < 4; ++nt)
#pragma unroll
    for (int r = 0; r < 8; ++r)
      sS[wid][r + 8 * half][nt * 16 + m] = o[nt][r];
  __syncthreads();
  if (lane < 16) {
    const float rinv = 1.0f / li;
    const int q = qBase + lane;
    const int b = bh >> 4, h = bh & 15;
    float* orow = out + (size_t)q * (BATCH * DIM) + (size_t)b * DIM + h * HEAD;
    for (int j = 0; j < 64; ++j) orow[j] = sS[wid][lane][j] * rinv;
  }
}

// ---------------------------------------------------------------------------
extern "C" void kernel_launch(void* const* d_in, const int* in_sizes, int n_in,
                              void* d_out, int out_size, void* d_ws,
                              size_t ws_size, hipStream_t stream) {
  (void)in_sizes; (void)n_in; (void)out_size; (void)ws_size;

  const float* K     = (const float*)d_in[0];
  const float* Q     = (const float*)d_in[1];
  const float* V     = (const float*)d_in[2];
  const float* WK    = (const float*)d_in[3];
  const float* bK    = (const float*)d_in[4];
  const float* WQ    = (const float*)d_in[5];
  const float* bQ    = (const float*)d_in[6];
  const float* WV    = (const float*)d_in[7];
  const float* bV    = (const float*)d_in[8];
  const float* gK    = (const float*)d_in[9];
  const float* betaK = (const float*)d_in[10];
  const float* gQ    = (const float*)d_in[11];
  const float* betaQ = (const float*)d_in[12];
  const float* gV    = (const float*)d_in[13];
  const float* betaV = (const float*)d_in[14];
  float* outp = (float*)d_out;

  // Workspace layout (bf16 buffers), total ~54 MB
  char* ws = (char*)d_ws;
  const size_t NX = (size_t)ROWS * DIM;
  const size_t NW = (size_t)DIM * DIM;
  const size_t NH = (size_t)BATCH * NHEAD * SEQ * HEAD;
  __bf16* Kb  = (__bf16*)(ws);
  __bf16* Qb  = Kb + NX;
  __bf16* Vb  = Qb + NX;
  __bf16* WKb = Vb + NX;
  __bf16* WQb = WKb + NW;
  __bf16* WVb = WQb + NW;
  __bf16* KeyN   = WVb + NW;     // [B,H,S,64]
  __bf16* QueryN = KeyN + NH;    // [B,H,S,64]
  __bf16* ValueT = QueryN + NH;  // [B,H,64,S]

  const int CT = 256;
  cvt_f32_bf16<<<dim3((unsigned)(NX / CT)), CT, 0, stream>>>(K, Kb, (int)NX);
  cvt_f32_bf16<<<dim3((unsigned)(NX / CT)), CT, 0, stream>>>(Q, Qb, (int)NX);
  cvt_f32_bf16<<<dim3((unsigned)(NX / CT)), CT, 0, stream>>>(V, Vb, (int)NX);
  cvt_f32_bf16<<<dim3((unsigned)(NW / CT)), CT, 0, stream>>>(WK, WKb, (int)NW);
  cvt_f32_bf16<<<dim3((unsigned)(NW / CT)), CT, 0, stream>>>(WQ, WQb, (int)NW);
  cvt_f32_bf16<<<dim3((unsigned)(NW / CT)), CT, 0, stream>>>(WV, WVb, (int)NW);

  dim3 pgrid(ROWS / 128, NHEAD);
  proj_ln_kernel<<<pgrid, 256, 0, stream>>>(Kb, WKb, bK, gK, betaK, KeyN,   0);
  proj_ln_kernel<<<pgrid, 256, 0, stream>>>(Qb, WQb, bQ, gQ, betaQ, QueryN, 0);
  proj_ln_kernel<<<pgrid, 256, 0, stream>>>(Vb, WVb, bV, gV, betaV, ValueT, 1);

  dim3 agrid(SEQ / 64, BATCH * NHEAD);
  attn_kernel<<<agrid, 128, 0, stream>>>(QueryN, KeyN, ValueT, outp);
}
